// GraphColorizer_41884521070630
// MI455X (gfx1250) — matrix operand order repaired and checked by reading, not compiled, not run on hardware
//
#include <hip/hip_runtime.h>
#include <hip/hip_bf16.h>
#include <math.h>

// ---------------------------------------------------------------------------
// Fused 3-layer MLP + masked softmax for MI455X (gfx1250, wave32, WMMA).
//  - GEMMs on v_wmma_f32_16x16x32_bf16 (fp32 accumulate)
//  - weights pre-swizzled into WMMA-B-fragment order in d_ws (coalesced 512B
//    bursts per fragment chunk, zero-padded so no clamping in the hot loop)
//  - all intermediates in LDS; 134KB/WG -> 2 workgroups per WGP (2 waves/SIMD)
// ---------------------------------------------------------------------------

typedef __attribute__((ext_vector_type(16))) __bf16 v16bf;
typedef __attribute__((ext_vector_type(8)))  __bf16 v8bf;
typedef __attribute__((ext_vector_type(4)))  __bf16 v4bf;
typedef __attribute__((ext_vector_type(8)))  float  v8f;

union ABf { v16bf v; v8bf h[2]; };

#define EMB        512
#define HID        400
#define NOUT       257        // N_COLORS + 1
#define ROWS_BLK   64         // rows per workgroup
#define NTHREADS   128        // 4 waves (wave32)
#define XS_STRIDE  520        // bf16 elems, padded (breaks mod-64 banking)
#define H2_STRIDE  424        // bf16 elems, K padded to 416
#define H2_KPAD    416
#define LG_STRIDE  276        // f32 elems

// LDS layout, two overlaid regions (total 137216 B -> 2 WGs per 320KB WGP):
//  R0 [0,66560):      xs [64][520] bf16;   later h2 [64][424] bf16 + rowM/rowI
//  R1 [66560,137216): h1 [64][520] bf16;   later lgs [64][276] f32
#define OFF_XS  0
#define OFF_H1  66560
#define OFF_LG  66560
#define OFF_H2  0
#define OFF_RM  54272         // float[64] (inside R0, past h2's 54272 bytes? no:
                              // h2 is dead before rowM is written, overlay is fine)
#define OFF_RI  54528
#define SMEM_BYTES 137216

// ---- swizzled bf16 weight workspace ---------------------------------------
// For weight matrix [NR][K] and tile (jt, kb), the WMMA B fragment
// (B[k][n] = W[n][k], n = jt*16 + (lane&15), k = kb*32 + 16c + 8*(lane>>4) + i)
// is stored as two contiguous 256-elem chunks:
//   dst[ ((jt*KB + kb)*2 + c)*256 + lane*8 + i ]
// Out-of-range n/k are stored as 0 (kills all clamping in the GEMM loop).
#define KB1 16
#define NT1 32                // 512/16 j-tiles
#define KB2 16
#define NT2 28                // ceil(400/16) rounded to 4-tile passes
#define KB3 13                // K=400 padded to 416
#define NT3 20                // ceil(257/16) rounded to 4-tile passes
#define SEG1 (NT1*KB1*512)    // 262144
#define SEG2 (NT2*KB2*512)    // 229376
#define SEG3 (NT3*KB3*512)    // 133120
#define CONV_TOTAL (SEG1+SEG2+SEG3)   // 624640 elems = 1.19 MB in d_ws

__device__ __forceinline__ void conv_seg(const float* __restrict__ W,
                                         int NR, int KORIG, int KBn,
                                         __bf16* __restrict__ dst, int e) {
    int within = e & 7;
    int lane   = (e >> 3) & 31;
    int rest   = e >> 8;
    int c      = rest & 1;
    rest >>= 1;
    int kb = rest % KBn;
    int jt = rest / KBn;
    int n = jt * 16 + (lane & 15);
    int k = kb * 32 + c * 16 + (lane >> 4) * 8 + within;
    float v = (n < NR && k < KORIG) ? W[n * KORIG + k] : 0.f;
    dst[e] = (__bf16)v;
}

__global__ void convert_weights_bf16(const float* __restrict__ W1,
                                     const float* __restrict__ W2,
                                     const float* __restrict__ W3,
                                     __bf16* __restrict__ ws) {
    int idx = blockIdx.x * 256 + threadIdx.x;
    if (idx < SEG1)             conv_seg(W1, EMB,  EMB, KB1, ws,             idx);
    else if (idx < SEG1 + SEG2) conv_seg(W2, HID,  EMB, KB2, ws + SEG1,      idx - SEG1);
    else if (idx < CONV_TOTAL)  conv_seg(W3, NOUT, HID, KB3, ws + SEG1+SEG2, idx - SEG1 - SEG2);
}

// One pass: accumulate 4 adjacent 16x16 output tiles over KBn*32 of K.
// A slab from LDS (two 16B chunks / lane), B from swizzled global bf16
// (two fully-coalesced 512B bursts per fragment).
__device__ __forceinline__ void mlp_pass(const __bf16* a0,   // LDS: (r0+l15)*stride + hlf*8
                                         const __bf16* wt,   // ws: jt0*KBn*512 + lane*8
                                         int KBn, v8f acc[4]) {
#pragma unroll 4
    for (int kb = 0; kb < KBn; ++kb) {
        ABf A;
        A.h[0] = *(const v8bf*)(a0 + kb * 32);
        A.h[1] = *(const v8bf*)(a0 + kb * 32 + 16);
#pragma unroll
        for (int t = 0; t < 4; ++t) {
            ABf B;
            B.h[0] = *(const v8bf*)(wt + (t * KBn + kb) * 512);
            B.h[1] = *(const v8bf*)(wt + (t * KBn + kb) * 512 + 256);
            acc[t] = __builtin_amdgcn_wmma_f32_16x16x32_bf16(
                         false, A.v, false, B.v, (short)0, acc[t], false, false);
        }
    }
}

__global__ __launch_bounds__(NTHREADS)
void mlp_fused(const float* __restrict__ x,
               const float* __restrict__ b1,
               const float* __restrict__ b2,
               const float* __restrict__ b3,
               const int*   __restrict__ nUsedPtr,
               const __bf16* __restrict__ wbf,
               float* __restrict__ out) {
    __shared__ __align__(16) char smem[SMEM_BYTES];
    __bf16* xs  = (__bf16*)(smem + OFF_XS);   // [64][520] bf16
    __bf16* h1  = (__bf16*)(smem + OFF_H1);   // [64][520] bf16
    float*  lgs = (float*) (smem + OFF_LG);   // [64][276] f32 (overlays h1)
    __bf16* h2  = (__bf16*)(smem + OFF_H2);   // [64][424] bf16 (overlays xs)
    float*  rowM = (float*)(smem + OFF_RM);
    float*  rowI = (float*)(smem + OFF_RI);

    const __bf16* W1b = wbf;
    const __bf16* W2b = wbf + SEG1;
    const __bf16* W3b = wbf + SEG1 + SEG2;

    const int tid  = threadIdx.x;
    const int lane = tid & 31;
    const int wave = tid >> 5;
    const int hlf  = lane >> 4;
    const int l15  = lane & 15;
    const int r0   = wave * 16;
    const long rowBase = (long)blockIdx.x * ROWS_BLK;

    // ---- stage x[64][512] fp32 -> bf16 LDS (coalesced float4 loads) ----
    for (int i = tid; i < ROWS_BLK * (EMB / 4); i += NTHREADS) {
        int r = i >> 7, c4 = i & 127;
        float4 f = ((const float4*)(x + (rowBase + r) * EMB))[c4];
        v4bf t;
        t[0] = (__bf16)f.x; t[1] = (__bf16)f.y;
        t[2] = (__bf16)f.z; t[3] = (__bf16)f.w;
        *(v4bf*)(xs + r * XS_STRIDE + c4 * 4) = t;
    }
    __syncthreads();

    // ---- Layer 1: h1 = leaky(x @ W1^T + b1)   [64,512] ----
    {
        const __bf16* a0 = xs + (r0 + l15) * XS_STRIDE + hlf * 8;
        for (int p = 0; p < 8; ++p) {
            v8f acc[4];
#pragma unroll
            for (int t = 0; t < 4; ++t)
#pragma unroll
                for (int v = 0; v < 8; ++v) acc[t][v] = 0.f;
            mlp_pass(a0, W1b + (long)(p * 4) * KB1 * 512 + lane * 8, KB1, acc);
#pragma unroll
            for (int t = 0; t < 4; ++t) {
                int col = (p * 4 + t) * 16 + l15;
                float bias = b1[col];
#pragma unroll
                for (int v = 0; v < 8; ++v) {
                    float val = acc[t][v] + bias;
                    val = val >= 0.f ? val : 0.01f * val;
                    h1[(r0 + hlf * 8 + v) * XS_STRIDE + col] = (__bf16)val;
                }
            }
        }
    }
    __syncthreads();

    // ---- Layer 2: h2 = leaky(h1 @ W2^T + b2)  [64,400], K-padded to 416 ----
    {
        const __bf16* a0 = h1 + (r0 + l15) * XS_STRIDE + hlf * 8;
        for (int p = 0; p < 7; ++p) {
            v8f acc[4];
#pragma unroll
            for (int t = 0; t < 4; ++t)
#pragma unroll
                for (int v = 0; v < 8; ++v) acc[t][v] = 0.f;
            mlp_pass(a0, W2b + (long)(p * 4) * KB2 * 512 + lane * 8, KB2, acc);
#pragma unroll
            for (int t = 0; t < 4; ++t) {
                int col = (p * 4 + t) * 16 + l15;
                if (col < H2_KPAD) {
                    float bias = (col < HID) ? b2[col] : 0.f;
#pragma unroll
                    for (int v = 0; v < 8; ++v) {
                        float val = acc[t][v] + bias;
                        val = val >= 0.f ? val : 0.01f * val;
                        __bf16 o = (col < HID) ? (__bf16)val : (__bf16)0.f;
                        h2[(r0 + hlf * 8 + v) * H2_STRIDE + col] = o;
                    }
                }
            }
        }
    }
    __syncthreads();

    // ---- Layer 3: logits = h2 @ W3^T + b3     [64,257] fp32 -> LDS ----
    {
        const __bf16* a0 = h2 + (r0 + l15) * H2_STRIDE + hlf * 8;
        for (int p = 0; p < 5; ++p) {
            v8f acc[4];
#pragma unroll
            for (int t = 0; t < 4; ++t)
#pragma unroll
                for (int v = 0; v < 8; ++v) acc[t][v] = 0.f;
            mlp_pass(a0, W3b + (long)(p * 4) * KB3 * 512 + lane * 8, KB3, acc);
#pragma unroll
            for (int t = 0; t < 4; ++t) {
                int col = (p * 4 + t) * 16 + l15;
                if (col < 272) {
                    float bias = (col < NOUT) ? b3[col] : 0.f;
#pragma unroll
                    for (int v = 0; v < 8; ++v)
                        lgs[(r0 + hlf * 8 + v) * LG_STRIDE + col] = acc[t][v] + bias;
                }
            }
        }
    }
    __syncthreads();

    // ---- masked softmax: valid cols are [0, n_used) and 256 ----
    const int n_used = *nUsedPtr;
    if (tid < ROWS_BLK) {
        const float* row = lgs + tid * LG_STRIDE;
        float m = row[256];                        // col 256 always valid
        for (int c = 0; c < 256; ++c)
            if (c < n_used) m = fmaxf(m, row[c]);
        float s = expf(row[256] - m);
        for (int c = 0; c < 256; ++c)
            if (c < n_used) s += expf(row[c] - m);
        rowM[tid] = m;
        rowI[tid] = 1.f / s;
    }
    __syncthreads();

    const long outBase = rowBase * NOUT;
    for (int i = tid; i < ROWS_BLK * NOUT; i += NTHREADS) {
        int r = i / NOUT;
        int c = i - r * NOUT;
        float val = 0.f;
        if (c < n_used || c == 256)
            val = expf(lgs[r * LG_STRIDE + c] - rowM[r]) * rowI[r];
        out[outBase + i] = val;
    }
}

extern "C" void kernel_launch(void* const* d_in, const int* in_sizes, int n_in,
                              void* d_out, int out_size, void* d_ws, size_t ws_size,
                              hipStream_t stream) {
    const float* x  = (const float*)d_in[0];
    const float* W1 = (const float*)d_in[1];
    const float* b1 = (const float*)d_in[2];
    const float* W2 = (const float*)d_in[3];
    const float* b2 = (const float*)d_in[4];
    const float* W3 = (const float*)d_in[5];
    const float* b3 = (const float*)d_in[6];
    const int*   nU = (const int*)d_in[7];
    __bf16* wbf = (__bf16*)d_ws;                 // needs >= 1.2 MB workspace
    float* out = (float*)d_out;

    const int N = in_sizes[0] / EMB;             // 131072

    convert_weights_bf16<<<(CONV_TOTAL + 255) / 256, 256, 0, stream>>>(W1, W2, W3, wbf);
    mlp_fused<<<N / ROWS_BLK, NTHREADS, 0, stream>>>(x, b1, b2, b3, nU, wbf, out);
}
